// Self_Cross_Attention_49761491092166
// MI455X (gfx1250) — compile-verified, matrix operand order checked by reading, not run on hardware
//
#include <hip/hip_runtime.h>

// ---------------- problem constants ----------------
#define EDIM 256
#define HNUM 8
#define DKL 32
#define NAQ 2048
#define NBK 2048
#define NKV (NAQ + NBK)   // 4096
#define BATCH 2

typedef __attribute__((ext_vector_type(16))) _Float16 v16h;
typedef __attribute__((ext_vector_type(8)))  _Float16 v8h;
typedef __attribute__((ext_vector_type(8)))  float    v8f;
typedef __attribute__((ext_vector_type(4)))  float    v4f;

__device__ __forceinline__ v8f wmma_f16(v16h a, v16h b, v8f c) {
  // D = A(16x32 f16) * B(32x16 f16) + C(16x16 f32)
  return __builtin_amdgcn_wmma_f32_16x16x32_f16(
      /*neg_a=*/false, a, /*neg_b=*/false, b,
      /*c_mod=*/(short)0, c, /*reuse_a=*/false, /*reuse_b=*/false);
}

__device__ __forceinline__ v8h cvt8(v4f a, v4f b) {
  v8h r;
  r[0] = (_Float16)a[0]; r[1] = (_Float16)a[1];
  r[2] = (_Float16)a[2]; r[3] = (_Float16)a[3];
  r[4] = (_Float16)b[0]; r[5] = (_Float16)b[1];
  r[6] = (_Float16)b[2]; r[7] = (_Float16)b[3];
  return r;
}

#define LSTR 48   // LDS row stride in halves (96B: keeps v16h reads 32B aligned)

// ---- tile staging helpers (256-thread blocks) ----
__device__ __forceinline__ void stageA_f32(const float* __restrict__ X, int m0,
                                           int kk, int tid, _Float16* As) {
  const int r  = tid >> 1;
  const int cb = (tid & 1) * 16;
  const float* src = X + (size_t)(m0 + r) * EDIM + kk + cb;
  v4f f0 = *(const v4f*)(src + 0);
  v4f f1 = *(const v4f*)(src + 4);
  v4f f2 = *(const v4f*)(src + 8);
  v4f f3 = *(const v4f*)(src + 12);
  _Float16* dst = As + r * LSTR + cb;
  *(v8h*)(dst + 0) = cvt8(f0, f1);
  *(v8h*)(dst + 8) = cvt8(f2, f3);
}

__device__ __forceinline__ void stageA_f16(const _Float16* __restrict__ Zin, int m0,
                                           int kk, int tid, _Float16* As) {
  const int r  = tid >> 1;
  const int cb = (tid & 1) * 16;
  v16h ad = *(const v16h*)(Zin + (size_t)(m0 + r) * EDIM + kk + cb);
  *(v16h*)(As + r * LSTR + cb) = ad;
}

__device__ __forceinline__ void stageB_f32(const float* __restrict__ W, int n0,
                                           int kk, int tid, _Float16* Bs) {
  const int r  = tid >> 2;
  const int cb = (tid & 3) * 8;
  const float* src = W + (size_t)(n0 + r) * EDIM + kk + cb;
  v4f f0 = *(const v4f*)(src + 0);
  v4f f1 = *(const v4f*)(src + 4);
  *(v8h*)(Bs + r * LSTR + cb) = cvt8(f0, f1);
}

// ============================================================
// Kernel 1: Y = X @ W^T + bias  (fp32 in, f16 out, head-split)
//   X: [B*rows, E] fp32 ; W: [E,E] fp32 ; out: [B,H,n_out_rows,DK] f16
//   grid (M/128, E/64), block 256 (8 waves, 4x2 wave grid, 32x32/wave)
//   double-buffered LDS: one barrier per k-step
// ============================================================
__global__ __launch_bounds__(256) void proj_heads_kernel(
    const float* __restrict__ X, const float* __restrict__ W,
    const float* __restrict__ bias, _Float16* __restrict__ out,
    int rows_per_batch, int n_out_rows, int n_off)
{
  __shared__ _Float16 As[2][128 * LSTR];
  __shared__ _Float16 Bs[2][64 * LSTR];

  const int tid  = threadIdx.x;
  const int lane = tid & 31;
  const int wave = tid >> 5;
  const int wm   = wave >> 1;        // 0..3
  const int wn   = wave & 1;         // 0..1
  const int m0   = blockIdx.x * 128;
  const int n0   = blockIdx.y * 64;
  const int lo   = lane & 15;
  const int hi   = lane >> 4;
  const int chunk = hi * 16;

  v8f acc[2][2];
  #pragma unroll
  for (int i = 0; i < 2; ++i)
    #pragma unroll
    for (int j = 0; j < 2; ++j) acc[i][j] = (v8f){};

  stageA_f32(X, m0, 0, tid, As[0]);
  stageB_f32(W, n0, 0, tid, Bs[0]);

  int buf = 0;
  for (int kk = 0; kk < EDIM; kk += 32) {
    __syncthreads();
    if (kk + 32 < EDIM) {
      stageA_f32(X, m0, kk + 32, tid, As[buf ^ 1]);
      stageB_f32(W, n0, kk + 32, tid, Bs[buf ^ 1]);
    }
    const _Float16* as = As[buf];
    const _Float16* bs = Bs[buf];

    const int arow = wm * 32 + lo;
    v16h a0 = *(const v16h*)(as + arow * LSTR + chunk);
    v16h a1 = *(const v16h*)(as + (arow + 16) * LSTR + chunk);
    const int bcol = wn * 32 + lo;
    v16h b0 = *(const v16h*)(bs + bcol * LSTR + chunk);
    v16h b1 = *(const v16h*)(bs + (bcol + 16) * LSTR + chunk);

    acc[0][0] = wmma_f16(a0, b0, acc[0][0]);
    acc[0][1] = wmma_f16(a0, b1, acc[0][1]);
    acc[1][0] = wmma_f16(a1, b0, acc[1][0]);
    acc[1][1] = wmma_f16(a1, b1, acc[1][1]);
    buf ^= 1;
  }

  // ---- epilogue: +bias, scatter into [B,H,n_out_rows,DK] f16 ----
  #pragma unroll
  for (int mi = 0; mi < 2; ++mi) {
    #pragma unroll
    for (int ni = 0; ni < 2; ++ni) {
      const int c = n0 + wn * 32 + ni * 16 + lo;
      const float bv = bias[c];
      const int hh = c >> 5;
      const int dk = c & 31;
      #pragma unroll
      for (int r = 0; r < 8; ++r) {
        const int m = m0 + wm * 32 + mi * 16 + r + hi * 8;
        const float v = acc[mi][ni][r] + bv;
        const int bb = m / rows_per_batch;
        const int n  = m % rows_per_batch;
        out[(((size_t)bb * HNUM + hh) * n_out_rows + n_off + n) * DKL + dk] =
            (_Float16)v;
      }
    }
  }
}

// ============================================================
// Kernel 2: flash attention.
//   Q: [B,H,NA,DK] f16 ; K,V: [B,H,NK,DK] f16 ; Z: [B,NA,E] f16
//   block = 128 threads (4 waves x 16 queries = 64 q / block)
//   grid  = (NA/64, B*H); double-buffered K/V tiles + prefetch
// ============================================================
__global__ __launch_bounds__(128) void flash_attn_kernel(
    const _Float16* __restrict__ Q, const _Float16* __restrict__ K,
    const _Float16* __restrict__ V, _Float16* __restrict__ Z)
{
  __shared__ _Float16 Ks[2][32 * LSTR];     // [key][dk]
  __shared__ _Float16 Vs[2][32 * LSTR];     // [dk][key]  (transposed)
  __shared__ _Float16 Ps[4 * 16 * 32];      // per-wave P staging

  const int tid  = threadIdx.x;
  const int lane = tid & 31;
  const int wave = tid >> 5;
  const int lo   = lane & 15;
  const int hi   = lane >> 4;
  const int chunk = hi * 16;

  const int bh = blockIdx.y;
  const int b  = bh >> 3;
  const int h  = bh & 7;
  const int q0 = blockIdx.x * 64 + wave * 16;

  // Q fragment: A-operand, 16 queries x 32 dk
  const _Float16* qbase = Q + ((size_t)bh * NAQ + q0) * DKL;
  v16h qfrag = *(const v16h*)(qbase + lo * DKL + chunk);

  const _Float16* kbase = K + (size_t)bh * NKV * DKL;
  const _Float16* vbase = V + (size_t)bh * NKV * DKL;

  // staging coordinates for this thread
  const int skey = tid >> 2;          // 0..31
  const int scb  = (tid & 3) * 8;     // 0,8,16,24

  v8f o0 = (v8f){}, o1 = (v8f){};
  float mrun[8], lrun[8];
  #pragma unroll
  for (int r = 0; r < 8; ++r) { mrun[r] = -1.0e30f; lrun[r] = 0.0f; }

  const float scale = 0.17677669529663687f;  // 1/sqrt(32)

  // prologue stage: tile 0
  {
    v8h kd = *(const v8h*)(kbase + (size_t)skey * DKL + scb);
    *(v8h*)(Ks[0] + skey * LSTR + scb) = kd;
    v8h vd = *(const v8h*)(vbase + (size_t)skey * DKL + scb);
    #pragma unroll
    for (int j = 0; j < 8; ++j)
      Vs[0][(scb + j) * LSTR + skey] = vd[j];
  }

  int buf = 0;
  for (int kt = 0; kt < NKV; kt += 32) {
    __syncthreads();
    // stage next tile into the other buffer (overlaps with WMMA below)
    if (kt + 32 < NKV) {
      v8h kd = *(const v8h*)(kbase + (size_t)(kt + 32 + skey) * DKL + scb);
      *(v8h*)(Ks[buf ^ 1] + skey * LSTR + scb) = kd;
      v8h vd = *(const v8h*)(vbase + (size_t)(kt + 32 + skey) * DKL + scb);
      #pragma unroll
      for (int j = 0; j < 8; ++j)
        Vs[buf ^ 1][(scb + j) * LSTR + skey] = vd[j];
    }
    // prefetch tile kt+64 into cache (global_prefetch_b8)
    if (kt + 64 < NKV) {
      __builtin_prefetch(kbase + (size_t)(kt + 64 + skey) * DKL + scb, 0, 0);
      __builtin_prefetch(vbase + (size_t)(kt + 64 + skey) * DKL + scb, 0, 0);
    }

    // ---- S = Q @ K^T : two 16x16 tiles over 32 keys ----
    const _Float16* ks = Ks[buf];
    const _Float16* vs = Vs[buf];
    v16h kf0 = *(const v16h*)(ks + lo * LSTR + chunk);
    v16h kf1 = *(const v16h*)(ks + (lo + 16) * LSTR + chunk);
    v8f zc = (v8f){};
    v8f s0 = wmma_f16(qfrag, kf0, zc);
    v8f s1 = wmma_f16(qfrag, kf1, zc);

    // ---- online softmax (rows live across 16-lane halves) ----
    _Float16* pw = Ps + wave * 512;
    #pragma unroll
    for (int r = 0; r < 8; ++r) {
      float a0 = s0[r] * scale;
      float a1 = s1[r] * scale;
      float mx = fmaxf(a0, a1);
      mx = fmaxf(mx, __shfl_xor(mx, 1, 32));
      mx = fmaxf(mx, __shfl_xor(mx, 2, 32));
      mx = fmaxf(mx, __shfl_xor(mx, 4, 32));
      mx = fmaxf(mx, __shfl_xor(mx, 8, 32));
      float mnew  = fmaxf(mrun[r], mx);
      float alpha = __expf(mrun[r] - mnew);
      float p0 = __expf(a0 - mnew);
      float p1 = __expf(a1 - mnew);
      float rs = p0 + p1;
      rs += __shfl_xor(rs, 1, 32);
      rs += __shfl_xor(rs, 2, 32);
      rs += __shfl_xor(rs, 4, 32);
      rs += __shfl_xor(rs, 8, 32);
      lrun[r] = lrun[r] * alpha + rs;
      mrun[r] = mnew;
      o0[r] *= alpha;
      o1[r] *= alpha;
      const int prow = r + hi * 8;
      pw[prow * 32 + lo]      = (_Float16)p0;
      pw[prow * 32 + lo + 16] = (_Float16)p1;
    }
    // wave-local LDS RAW fence before reading P as an A-fragment
    asm volatile("s_wait_dscnt 0" ::: "memory");
    __builtin_amdgcn_wave_barrier();

    v16h pfrag = *(const v16h*)(pw + lo * 32 + chunk);
    v16h vf0 = *(const v16h*)(vs + lo * LSTR + chunk);          // dk 0..15
    v16h vf1 = *(const v16h*)(vs + (lo + 16) * LSTR + chunk);   // dk 16..31
    o0 = wmma_f16(pfrag, vf0, o0);
    o1 = wmma_f16(pfrag, vf1, o1);
    buf ^= 1;
  }

  // ---- normalize + write z in [B,NA,E] f16 ----
  #pragma unroll
  for (int r = 0; r < 8; ++r) {
    const float inv = 1.0f / lrun[r];
    const int gq = q0 + r + hi * 8;
    const size_t base = ((size_t)b * NAQ + gq) * EDIM + h * DKL;
    Z[base + lo]      = (_Float16)(o0[r] * inv);
    Z[base + lo + 16] = (_Float16)(o1[r] * inv);
  }
}

// ============================================================
// Kernel 3: out = z @ w^T + b + x_a   (f16 A, fp32 W/res, fp32 out)
//   grid (M/128, E/64), block 256; double-buffered LDS
// ============================================================
__global__ __launch_bounds__(256) void out_proj_kernel(
    const _Float16* __restrict__ Zin, const float* __restrict__ W,
    const float* __restrict__ bias, const float* __restrict__ Xres,
    float* __restrict__ out)
{
  __shared__ _Float16 As[2][128 * LSTR];
  __shared__ _Float16 Bs[2][64 * LSTR];

  const int tid  = threadIdx.x;
  const int lane = tid & 31;
  const int wave = tid >> 5;
  const int wm   = wave >> 1;
  const int wn   = wave & 1;
  const int m0   = blockIdx.x * 128;
  const int n0   = blockIdx.y * 64;
  const int lo   = lane & 15;
  const int hi   = lane >> 4;
  const int chunk = hi * 16;

  v8f acc[2][2];
  #pragma unroll
  for (int i = 0; i < 2; ++i)
    #pragma unroll
    for (int j = 0; j < 2; ++j) acc[i][j] = (v8f){};

  stageA_f16(Zin, m0, 0, tid, As[0]);
  stageB_f32(W, n0, 0, tid, Bs[0]);

  int buf = 0;
  for (int kk = 0; kk < EDIM; kk += 32) {
    __syncthreads();
    if (kk + 32 < EDIM) {
      stageA_f16(Zin, m0, kk + 32, tid, As[buf ^ 1]);
      stageB_f32(W, n0, kk + 32, tid, Bs[buf ^ 1]);
    }
    const _Float16* as = As[buf];
    const _Float16* bs = Bs[buf];

    const int arow = wm * 32 + lo;
    v16h a0 = *(const v16h*)(as + arow * LSTR + chunk);
    v16h a1 = *(const v16h*)(as + (arow + 16) * LSTR + chunk);
    const int bcol = wn * 32 + lo;
    v16h b0 = *(const v16h*)(bs + bcol * LSTR + chunk);
    v16h b1 = *(const v16h*)(bs + (bcol + 16) * LSTR + chunk);

    acc[0][0] = wmma_f16(a0, b0, acc[0][0]);
    acc[0][1] = wmma_f16(a0, b1, acc[0][1]);
    acc[1][0] = wmma_f16(a1, b0, acc[1][0]);
    acc[1][1] = wmma_f16(a1, b1, acc[1][1]);
    buf ^= 1;
  }

  #pragma unroll
  for (int mi = 0; mi < 2; ++mi) {
    #pragma unroll
    for (int ni = 0; ni < 2; ++ni) {
      const int c = n0 + wn * 32 + ni * 16 + lo;
      const float bv = bias[c];
      #pragma unroll
      for (int r = 0; r < 8; ++r) {
        const int m = m0 + wm * 32 + mi * 16 + r + hi * 8;
        const size_t idx = (size_t)m * EDIM + c;
        out[idx] = acc[mi][ni][r] + bv + Xres[idx];
      }
    }
  }
}

// ============================================================
// launch
// ============================================================
extern "C" void kernel_launch(void* const* d_in, const int* in_sizes, int n_in,
                              void* d_out, int out_size, void* d_ws, size_t ws_size,
                              hipStream_t stream) {
  const float* x_a    = (const float*)d_in[0];
  const float* x_b    = (const float*)d_in[1];
  const float* wq_w   = (const float*)d_in[2];
  const float* wq_b   = (const float*)d_in[3];
  const float* wk_a_w = (const float*)d_in[4];
  const float* wk_a_b = (const float*)d_in[5];
  const float* wk_b_w = (const float*)d_in[6];
  const float* wk_b_b = (const float*)d_in[7];
  const float* wv_a_w = (const float*)d_in[8];
  const float* wv_a_b = (const float*)d_in[9];
  const float* wv_b_w = (const float*)d_in[10];
  const float* wv_b_b = (const float*)d_in[11];
  const float* w_w    = (const float*)d_in[12];
  const float* w_b    = (const float*)d_in[13];

  char* ws = (char*)d_ws;
  _Float16* qws = (_Float16*)(ws);                          // B*H*NA*DK  f16 = 2 MB
  _Float16* kws = (_Float16*)(ws + (size_t)(2  << 20));     // B*H*NK*DK  f16 = 4 MB
  _Float16* vws = (_Float16*)(ws + (size_t)(6  << 20));     // B*H*NK*DK  f16 = 4 MB
  _Float16* zws = (_Float16*)(ws + (size_t)(10 << 20));     // B*NA*E     f16 = 2 MB

  dim3 gp(32, 4);   // M=4096 /128, N=256 /64
  dim3 bp(256);

  // projections into head-split f16 layouts (K/V of the two sources concat via n_off)
  proj_heads_kernel<<<gp, bp, 0, stream>>>(x_a, wq_w,   wq_b,   qws, NAQ, NAQ, 0);
  proj_heads_kernel<<<gp, bp, 0, stream>>>(x_a, wk_a_w, wk_a_b, kws, NAQ, NKV, 0);
  proj_heads_kernel<<<gp, bp, 0, stream>>>(x_b, wk_b_w, wk_b_b, kws, NBK, NKV, NAQ);
  proj_heads_kernel<<<gp, bp, 0, stream>>>(x_a, wv_a_w, wv_a_b, vws, NAQ, NKV, 0);
  proj_heads_kernel<<<gp, bp, 0, stream>>>(x_b, wv_b_w, wv_b_b, vws, NBK, NKV, NAQ);

  // flash attention
  flash_attn_kernel<<<dim3(NAQ / 64, BATCH * HNUM), 128, 0, stream>>>(qws, kws, vws, zws);

  // output projection + residual
  out_proj_kernel<<<dim3(32, 4), 256, 0, stream>>>(zws, w_w, w_b, x_a, (float*)d_out);
}
